// Transformer_24850680775132
// MI455X (gfx1250) — matrix operand order repaired and checked by reading, not compiled, not run on hardware
//
#include <hip/hip_runtime.h>
#include <hip/hip_bf16.h>

// ---------------------------------------------------------------------------
// Model constants
// ---------------------------------------------------------------------------
#define TB   8        // batch
#define TS   512      // seq len
#define TD   512      // model dim
#define TH   8        // heads
#define TDH  64       // head dim
#define TFF  2048     // ffn dim
#define TNL  6        // layers
#define TCLS 10
#define TT   (TB*TS)         // 4096 tokens
#define TBH  (TB*TH)         // 64 attention batches
#define ATTL ((long)TBH*TS*TS) // 16,777,216 elems per attention stack layer

typedef __attribute__((ext_vector_type(16))) _Float16 v16h;
typedef __attribute__((ext_vector_type(8)))  _Float16 v8h;
typedef __attribute__((ext_vector_type(8)))  float    v8f;

// ---------------------------------------------------------------------------
// WMMA GEMM:  C[M,N] = act( alpha * A[M,K](f16) @ Wt[N,K](f16).T + bias + resid )
//   Wt is row-major N x K  (i.e. torch-style weight, computing x @ W.T)
//   Block = 256 threads = 8 waves, block tile 128x128, wave tile 32x64,
//   K-step 32 -> 8 v_wmma_f32_16x16x32_f16 per iteration.
// ---------------------------------------------------------------------------
__global__ __launch_bounds__(256)
void gemm_f16_wmma(const _Float16* __restrict__ A,
                   const _Float16* __restrict__ Wt,
                   const float* __restrict__ bias,
                   const float* __restrict__ resid,
                   float* __restrict__ Cf,
                   _Float16* __restrict__ Ch,
                   int M, int N, int K,
                   long strideA, long strideW, long strideC,
                   float alpha, int relu)
{
    const int batch = blockIdx.z;
    A  += (long)batch * strideA;
    Wt += (long)batch * strideW;

    const int tid    = threadIdx.x;
    const int wid    = tid >> 5;          // wave 0..7
    const int lane   = tid & 31;
    const int laneLo = lane & 15;
    const int hi     = lane >> 4;         // 0 | 1

    const int waveM = wid >> 1;           // 0..3
    const int waveN = wid & 1;            // 0..1
    const int wm0 = blockIdx.x * 128 + waveM * 32;   // 2 M-tiles of 16
    const int wn0 = blockIdx.y * 128 + waveN * 64;   // 4 N-tiles of 16

    v8f acc[2][4];
    const v8f vzero = {0.f,0.f,0.f,0.f,0.f,0.f,0.f,0.f};
#pragma unroll
    for (int i = 0; i < 2; ++i)
#pragma unroll
        for (int j = 0; j < 4; ++j) acc[i][j] = vzero;

    // clamped per-lane row indices (keeps all loads in-bounds)
    int arow[2], wrow[4];
#pragma unroll
    for (int i = 0; i < 2; ++i) { int r = wm0 + i*16 + laneLo; arow[i] = (r < M) ? r : (M-1); }
#pragma unroll
    for (int j = 0; j < 4; ++j) { int r = wn0 + j*16 + laneLo; wrow[j] = (r < N) ? r : (N-1); }

    for (int k0 = 0; k0 < K; k0 += 32) {
        // A fragment (16x32 f16): lanes 0-15 K=0..15, lanes 16-31 K=16..31
        v16h afrag[2];
#pragma unroll
        for (int i = 0; i < 2; ++i) {
            const _Float16* p = A + (long)arow[i]*K + k0 + hi*16;
            union { v16h v; v8h h[2]; } u;
            u.h[0] = *(const v8h*)(p);
            u.h[1] = *(const v8h*)(p + 8);
            afrag[i] = u.v;
        }
        // B fragment (32x16 f16): lane n<16 -> K=k0..k0+15 of row n; lane n+16 -> K=k0+16..k0+31
        v16h bfrag[4];
#pragma unroll
        for (int j = 0; j < 4; ++j) {
            const _Float16* p = Wt + (long)wrow[j]*K + k0 + hi*16;
            bfrag[j] = *(const v16h*)(p);
        }
#pragma unroll
        for (int i = 0; i < 2; ++i)
#pragma unroll
            for (int j = 0; j < 4; ++j)
                acc[i][j] = __builtin_amdgcn_wmma_f32_16x16x32_f16(
                    false, afrag[i], false, bfrag[j],
                    (short)0, acc[i][j], false, false);
    }

    // D layout: VGPR r -> row (r + 8*hi), col = laneLo
#pragma unroll
    for (int j = 0; j < 4; ++j) {
        const int col = wn0 + j*16 + laneLo;
        if (col >= N) continue;
        const float bv = bias ? bias[col] : 0.f;
#pragma unroll
        for (int i = 0; i < 2; ++i) {
            const int rbase = wm0 + i*16 + hi*8;
#pragma unroll
            for (int r = 0; r < 8; ++r) {
                const int row = rbase + r;
                if (row >= M) continue;
                float v = alpha * acc[i][j][r] + bv;
                if (resid) v += resid[(long)row * N + col];
                if (relu)  v = fmaxf(v, 0.f);
                const long ci = (long)batch * strideC + (long)row * N + col;
                if (Cf) Cf[ci] = v;
                if (Ch) Ch[ci] = (_Float16)v;
            }
        }
    }
}

// ---------------------------------------------------------------------------
// Row softmax over 512 (in-place f32 in d_out + f16 copy for attn@V)
// ---------------------------------------------------------------------------
__global__ __launch_bounds__(128)
void softmax512_kernel(float* __restrict__ x, _Float16* __restrict__ xh)
{
    const long row = blockIdx.x;
    float*     p  = x  + row * 512;
    _Float16*  ph = xh + row * 512;
    const int tid = threadIdx.x;
    __shared__ float red[128];

    float v[4]; float mx = -3.4e38f;
#pragma unroll
    for (int i = 0; i < 4; ++i) { v[i] = p[tid + i*128]; mx = fmaxf(mx, v[i]); }
    red[tid] = mx; __syncthreads();
    for (int s = 64; s > 0; s >>= 1) { if (tid < s) red[tid] = fmaxf(red[tid], red[tid+s]); __syncthreads(); }
    mx = red[0]; __syncthreads();

    float sum = 0.f;
#pragma unroll
    for (int i = 0; i < 4; ++i) { v[i] = __expf(v[i] - mx); sum += v[i]; }
    red[tid] = sum; __syncthreads();
    for (int s = 64; s > 0; s >>= 1) { if (tid < s) red[tid] += red[tid+s]; __syncthreads(); }
    const float inv = 1.f / red[0];
#pragma unroll
    for (int i = 0; i < 4; ++i) {
        const float o = v[i] * inv;
        p[tid + i*128]  = o;
        ph[tid + i*128] = (_Float16)o;
    }
}

// ---------------------------------------------------------------------------
// Row layernorm over D=512, writes f32 + f16
// ---------------------------------------------------------------------------
__global__ __launch_bounds__(128)
void layernorm512_kernel(const float* __restrict__ x,
                         const float* __restrict__ g, const float* __restrict__ b,
                         float* __restrict__ of, _Float16* __restrict__ oh)
{
    const long row = blockIdx.x;
    const float* p = x + row * 512;
    const int tid = threadIdx.x;
    __shared__ float red[128];

    float v[4]; float s = 0.f;
#pragma unroll
    for (int i = 0; i < 4; ++i) { v[i] = p[tid + i*128]; s += v[i]; }
    red[tid] = s; __syncthreads();
    for (int t = 64; t > 0; t >>= 1) { if (tid < t) red[tid] += red[tid+t]; __syncthreads(); }
    const float mean = red[0] * (1.f/512.f); __syncthreads();

    float vs = 0.f;
#pragma unroll
    for (int i = 0; i < 4; ++i) { const float d = v[i] - mean; vs += d*d; }
    red[tid] = vs; __syncthreads();
    for (int t = 64; t > 0; t >>= 1) { if (tid < t) red[tid] += red[tid+t]; __syncthreads(); }
    const float inv = rsqrtf(red[0] * (1.f/512.f) + 1e-5f);

#pragma unroll
    for (int i = 0; i < 4; ++i) {
        const int c = tid + i*128;
        const float o = (v[i] - mean) * inv * g[c] + b[c];
        of[row*512 + c] = o;
        oh[row*512 + c] = (_Float16)o;
    }
}

// ---------------------------------------------------------------------------
// token embedding + sinusoidal positional encoding -> f32 + f16
// ---------------------------------------------------------------------------
__global__ void embed_pe_kernel(const int* __restrict__ tok, const float* __restrict__ emb,
                                float* __restrict__ of, _Float16* __restrict__ oh, long n)
{
    const long idx = (long)blockIdx.x * 256 + threadIdx.x;
    if (idx >= n) return;
    const int  d  = (int)(idx & 511);
    const long ts = idx >> 9;               // b*S + s
    const int  s  = (int)(ts & 511);
    const int  t  = tok[ts];
    const float ang = (float)s * powf(10000.f, -(float)((d >> 1) << 1) / 512.f);
    const float pe  = (d & 1) ? cosf(ang) : sinf(ang);
    const float v = emb[(long)t * 512 + d] + pe;
    of[idx] = v;
    oh[idx] = (_Float16)v;
}

// ---------------------------------------------------------------------------
// V-head transpose: [64][512][64] -> [64][64][512]  (f16)
// ---------------------------------------------------------------------------
__global__ void transpose_v_kernel(const _Float16* __restrict__ src, _Float16* __restrict__ dst)
{
    const long idx = (long)blockIdx.x * 256 + threadIdx.x;
    if (idx >= (long)TBH * TS * TDH) return;
    const int  d  = (int)(idx & 63);
    const long r  = idx >> 6;
    const int  u  = (int)(r & 511);
    const int  bh = (int)(r >> 9);
    dst[(long)bh * 32768 + (long)d * 512 + u] = src[idx];
}

// ---------------------------------------------------------------------------
// f32 -> f16 cast (weights)
// ---------------------------------------------------------------------------
__global__ void cast_h_kernel(const float* __restrict__ s, _Float16* __restrict__ d, long n)
{
    const long i = (long)blockIdx.x * 256 + threadIdx.x;
    if (i < n) d[i] = (_Float16)s[i];
}

// ---------------------------------------------------------------------------
// final classifier: out[b][c] = dec[b,:] . fc_W[c,:] + fc_b[c]   (K = 262144)
// ---------------------------------------------------------------------------
__global__ __launch_bounds__(256)
void fc_kernel(const float* __restrict__ x, const float* __restrict__ W,
               const float* __restrict__ bias, float* __restrict__ out)
{
    const int bi = blockIdx.x;   // 0..7
    const int c  = blockIdx.y;   // 0..9
    const long K = (long)TS * TD;
    const float* xp = x + (long)bi * K;
    const float* wp = W + (long)c * K;
    float s = 0.f;
    for (long k = threadIdx.x; k < K; k += 256) s += xp[k] * wp[k];
    __shared__ float red[256];
    red[threadIdx.x] = s; __syncthreads();
    for (int t = 128; t > 0; t >>= 1) { if (threadIdx.x < t) red[threadIdx.x] += red[threadIdx.x + t]; __syncthreads(); }
    if (threadIdx.x == 0) out[bi * TCLS + c] = red[0] + bias[c];
}

// ===========================================================================
// Host orchestration
// ===========================================================================
struct WS {
    _Float16 *eWq,*eWk,*eWv,*eWo,*eW1,*eW2;
    _Float16 *dWq,*dWk,*dWv,*dWo,*dW1,*dW2;
    float *Xf;  _Float16 *Xh;
    float *decf; _Float16 *dech;
    float *encf; _Float16 *ench;
    float *Af;  _Float16 *Ah;    // mha out #1
    float *Bf;  _Float16 *Bh;    // mha out #2
    float *t1;                    // pre-LN temp
    _Float16 *Qp,*Kp,*Vp,*Vt,*ctx;
    _Float16 *attn;
    _Float16 *hff;
};

static inline void launch_gemm(hipStream_t st, const _Float16* A, const _Float16* Wt,
                               const float* bias, const float* resid,
                               float* Cf, _Float16* Ch,
                               int M, int N, int K, int batch,
                               long sA, long sW, long sC, float alpha, int relu)
{
    dim3 grid((M + 127) / 128, (N + 127) / 128, batch);
    gemm_f16_wmma<<<grid, 256, 0, st>>>(A, Wt, bias, resid, Cf, Ch,
                                        M, N, K, sA, sW, sC, alpha, relu);
}

static void run_mha(hipStream_t st, const WS& w,
                    const _Float16* q_h, const float* q_f, const _Float16* kv_h,
                    const _Float16* Wq, const float* bq,
                    const _Float16* Wk, const float* bk,
                    const _Float16* Wv, const float* bv,
                    const _Float16* Wo, const float* bo,
                    const float* g, const float* bb,
                    float* attn_out, float* of, _Float16* oh)
{
    // projections (f16 outputs, fed to attention WMMA)
    launch_gemm(st, q_h,  Wq, bq, nullptr, nullptr, w.Qp, TT, TD, TD, 1, 0,0,0, 1.f, 0);
    launch_gemm(st, kv_h, Wk, bk, nullptr, nullptr, w.Kp, TT, TD, TD, 1, 0,0,0, 1.f, 0);
    launch_gemm(st, kv_h, Wv, bv, nullptr, nullptr, w.Vp, TT, TD, TD, 1, 0,0,0, 1.f, 0);
    // scaled logits straight into d_out slice (batched over 64 heads)
    launch_gemm(st, w.Qp, w.Kp, nullptr, nullptr, attn_out, nullptr,
                TS, TS, TDH, TBH, (long)TS*TDH, (long)TS*TDH, (long)TS*TS, 0.125f, 0);
    softmax512_kernel<<<TBH*TS, 128, 0, st>>>(attn_out, w.attn);
    transpose_v_kernel<<<(TBH*TS*TDH + 255)/256, 256, 0, st>>>(w.Vp, w.Vt);
    // ctx = attn @ V  (Vt is N=64 x K=512 row-major -> same W.T convention)
    launch_gemm(st, w.attn, w.Vt, nullptr, nullptr, nullptr, w.ctx,
                TS, TDH, TS, TBH, (long)TS*TS, (long)TDH*TS, (long)TS*TDH, 1.f, 0);
    // out = LN(ctx @ Wo.T + bo + q)
    launch_gemm(st, w.ctx, Wo, bo, q_f, w.t1, nullptr, TT, TD, TD, 1, 0,0,0, 1.f, 0);
    layernorm512_kernel<<<TT, 128, 0, st>>>(w.t1, g, bb, of, oh);
}

static void run_ffn(hipStream_t st, const WS& w, const _Float16* in_h, const float* in_f,
                    const _Float16* W1, const float* b1,
                    const _Float16* W2, const float* b2,
                    const float* g, const float* bb, float* of, _Float16* oh)
{
    launch_gemm(st, in_h, W1, b1, nullptr, nullptr, w.hff, TT, TFF, TD, 1, 0,0,0, 1.f, 1);
    launch_gemm(st, w.hff, W2, b2, in_f, w.t1, nullptr, TT, TD, TFF, 1, 0,0,0, 1.f, 0);
    layernorm512_kernel<<<TT, 128, 0, st>>>(w.t1, g, bb, of, oh);
}

extern "C" void kernel_launch(void* const* d_in, const int* in_sizes, int n_in,
                              void* d_out, int out_size, void* d_ws, size_t ws_size,
                              hipStream_t stream)
{
    (void)in_sizes; (void)n_in; (void)out_size; (void)ws_size;

    const int*   x_tok = (const int*)d_in[0];
    const int*   y_tok = (const int*)d_in[1];
    const float* emb_x = (const float*)d_in[2];
    const float* emb_y = (const float*)d_in[3];
    const float* fc_W  = (const float*)d_in[4];
    const float* fc_b  = (const float*)d_in[5];
    const int E = 6, Dd = 22;   // enc / dec input bases

    float* out      = (float*)d_out;
    float* out_enc  = out + TB * TCLS;
    float* out_self = out_enc  + TNL * ATTL;
    float* out_ctx  = out_self + TNL * ATTL;

    // ---- workspace partition ----
    char* base = (char*)d_ws;
    size_t off = 0;
    auto alloc = [&](size_t bytes) -> void* {
        void* p = base + off;
        off = (off + bytes + 255) & ~(size_t)255;
        return p;
    };
    const size_t WQKVO = (size_t)TNL * TD * TD;    // halfs
    const size_t WFFN  = (size_t)TNL * TFF * TD;   // halfs
    const size_t ACT   = (size_t)TT * TD;          // elems

    WS w;
    w.eWq = (_Float16*)alloc(WQKVO*2); w.eWk = (_Float16*)alloc(WQKVO*2);
    w.eWv = (_Float16*)alloc(WQKVO*2); w.eWo = (_Float16*)alloc(WQKVO*2);
    w.eW1 = (_Float16*)alloc(WFFN*2);  w.eW2 = (_Float16*)alloc(WFFN*2);
    w.dWq = (_Float16*)alloc(WQKVO*2); w.dWk = (_Float16*)alloc(WQKVO*2);
    w.dWv = (_Float16*)alloc(WQKVO*2); w.dWo = (_Float16*)alloc(WQKVO*2);
    w.dW1 = (_Float16*)alloc(WFFN*2);  w.dW2 = (_Float16*)alloc(WFFN*2);
    w.Xf   = (float*)alloc(ACT*4);     w.Xh   = (_Float16*)alloc(ACT*2);
    w.decf = (float*)alloc(ACT*4);     w.dech = (_Float16*)alloc(ACT*2);
    w.encf = (float*)alloc(ACT*4);     w.ench = (_Float16*)alloc(ACT*2);
    w.Af   = (float*)alloc(ACT*4);     w.Ah   = (_Float16*)alloc(ACT*2);
    w.Bf   = (float*)alloc(ACT*4);     w.Bh   = (_Float16*)alloc(ACT*2);
    w.t1   = (float*)alloc(ACT*4);
    w.Qp   = (_Float16*)alloc(ACT*2);  w.Kp   = (_Float16*)alloc(ACT*2);
    w.Vp   = (_Float16*)alloc(ACT*2);  w.Vt   = (_Float16*)alloc(ACT*2);
    w.ctx  = (_Float16*)alloc(ACT*2);
    w.attn = (_Float16*)alloc((size_t)ATTL*2);
    w.hff  = (_Float16*)alloc((size_t)TT*TFF*2);

    // ---- weight f32 -> f16 casts ----
    auto cast = [&](const void* s, _Float16* d, long n) {
        cast_h_kernel<<<(unsigned)((n + 255) / 256), 256, 0, stream>>>((const float*)s, d, n);
    };
    cast(d_in[E+0],  w.eWq, WQKVO); cast(d_in[E+2],  w.eWk, WQKVO);
    cast(d_in[E+4],  w.eWv, WQKVO); cast(d_in[E+6],  w.eWo, WQKVO);
    cast(d_in[E+10], w.eW1, WFFN);  cast(d_in[E+12], w.eW2, WFFN);
    cast(d_in[Dd+0],  w.dWq, WQKVO); cast(d_in[Dd+2],  w.dWk, WQKVO);
    cast(d_in[Dd+4],  w.dWv, WQKVO); cast(d_in[Dd+6],  w.dWo, WQKVO);
    cast(d_in[Dd+10], w.dW1, WFFN);  cast(d_in[Dd+12], w.dW2, WFFN);

    // ---- embeddings + positional encoding ----
    embed_pe_kernel<<<(unsigned)((ACT + 255) / 256), 256, 0, stream>>>(x_tok, emb_x, w.Xf, w.Xh, (long)ACT);
    embed_pe_kernel<<<(unsigned)((ACT + 255) / 256), 256, 0, stream>>>(y_tok, emb_y, w.decf, w.dech, (long)ACT);

    // ---- encoder (faithful bug: every layer consumes the embedding) ----
    for (int l = 0; l < TNL; ++l) {
        const long wo = (long)l * TD * TD, fo1 = (long)l * TFF * TD, fo2 = (long)l * TD * TFF;
        run_mha(stream, w, w.Xh, w.Xf, w.Xh,
                w.eWq + wo, (const float*)d_in[E+1] + l*TD,
                w.eWk + wo, (const float*)d_in[E+3] + l*TD,
                w.eWv + wo, (const float*)d_in[E+5] + l*TD,
                w.eWo + wo, (const float*)d_in[E+7] + l*TD,
                (const float*)d_in[E+8] + l*TD, (const float*)d_in[E+9] + l*TD,
                out_enc + (long)l * ATTL, w.Af, w.Ah);
        run_ffn(stream, w, w.Ah, w.Af,
                w.eW1 + fo1, (const float*)d_in[E+11] + l*TFF,
                w.eW2 + fo2, (const float*)d_in[E+13] + l*TD,
                (const float*)d_in[E+14] + l*TD, (const float*)d_in[E+15] + l*TD,
                w.encf, w.ench);
    }

    // ---- decoder (chained; one shared MHA weight set per layer) ----
    for (int l = 0; l < TNL; ++l) {
        const long wo = (long)l * TD * TD, fo1 = (long)l * TFF * TD, fo2 = (long)l * TD * TFF;
        const _Float16 *Wq = w.dWq + wo, *Wk = w.dWk + wo, *Wv = w.dWv + wo, *Wo = w.dWo + wo;
        const float *bq = (const float*)d_in[Dd+1] + l*TD, *bk = (const float*)d_in[Dd+3] + l*TD;
        const float *bv = (const float*)d_in[Dd+5] + l*TD, *bo = (const float*)d_in[Dd+7] + l*TD;
        const float *g1 = (const float*)d_in[Dd+8] + l*TD, *b1l = (const float*)d_in[Dd+9] + l*TD;
        // self-attention
        run_mha(stream, w, w.dech, w.decf, w.dech,
                Wq, bq, Wk, bk, Wv, bv, Wo, bo, g1, b1l,
                out_self + (long)l * ATTL, w.Af, w.Ah);
        // cross-attention (same weights, K/V from encoder output)
        run_mha(stream, w, w.Ah, w.Af, w.ench,
                Wq, bq, Wk, bk, Wv, bv, Wo, bo, g1, b1l,
                out_ctx + (long)l * ATTL, w.Bf, w.Bh);
        // FFN -> next decoder state
        run_ffn(stream, w, w.Bh, w.Bf,
                w.dW1 + fo1, (const float*)d_in[Dd+11] + l*TFF,
                w.dW2 + fo2, (const float*)d_in[Dd+13] + l*TD,
                (const float*)d_in[Dd+14] + l*TD, (const float*)d_in[Dd+15] + l*TD,
                w.decf, w.dech);
    }

    // ---- final classifier ----
    fc_kernel<<<dim3(TB, TCLS), 256, 0, stream>>>(w.decf, fc_W, fc_b, out);
}
